// Sequence_50087908606395
// MI455X (gfx1250) — matrix operand order repaired and checked by reading, not compiled, not run on hardware
//
#include <hip/hip_runtime.h>

// ---------------------------------------------------------------------------
// 2-layer LSTM (T=512, B=256, F=510) + linear head, bf16 WMMA on gfx1250.
//
// Layout decisions:
//   Fp  = 512  : padded feature dim (K halves and h/c row stride)
//   K   = 1024 : concat(x_or_h1 [0..511] , h_prev [512..1023])
//   Wp  = [4 gates][512 cols][1024 K] bf16, zero padded
//   bias= [4][512] f32 = b_ih + b_hh, zero padded
// Padded columns stay exactly 0 through the recurrence (zero W, zero bias,
// zero init state => gates=0 => c_new=0.5*0+0.5*0=0, h_new=0.5*tanh(0)=0).
// ---------------------------------------------------------------------------

#define F_DIM 510
#define FP    512
#define BATCH 256
#define TSTEP 512
#define KDIM  1024

typedef __attribute__((ext_vector_type(16))) __bf16 v16bf;
typedef __attribute__((ext_vector_type(8)))  __bf16 bf16x8;
typedef __attribute__((ext_vector_type(8)))  float  v8f;

__device__ __forceinline__ float sigmoidf(float x) {
  return 1.0f / (1.0f + __expf(-x));
}

__device__ __forceinline__ v16bf load_frag16(const __bf16* p0, const __bf16* p1) {
  bf16x8 lo = *(const bf16x8*)p0;   // 16-byte load
  bf16x8 hi = *(const bf16x8*)p1;   // 16-byte load
  return __builtin_shufflevector(lo, hi, 0,1,2,3,4,5,6,7,8,9,10,11,12,13,14,15);
}

// ---- one-time packing kernels ---------------------------------------------

// x [T*B, 510] f32 -> xb [T*B, 512] bf16 (zero padded)
__global__ void conv_x_kernel(const float* __restrict__ x, __bf16* __restrict__ xb) {
  size_t idx = (size_t)blockIdx.x * blockDim.x + threadIdx.x; // < T*B*512
  int    k   = (int)(idx & (FP - 1));
  size_t row = idx >> 9;
  float  v   = (k < F_DIM) ? x[row * F_DIM + k] : 0.0f;
  xb[idx] = (__bf16)v;
}

// w_ih,w_hh [2040,510] f32 -> Wp [4][512][1024] bf16
__global__ void pack_w_kernel(const float* __restrict__ w_ih,
                              const float* __restrict__ w_hh,
                              __bf16* __restrict__ Wp) {
  size_t idx = (size_t)blockIdx.x * blockDim.x + threadIdx.x; // < 4*512*1024
  int k = (int)(idx & (KDIM - 1));
  int r = (int)((idx >> 10) & (FP - 1));
  int g = (int)(idx >> 19);
  float v = 0.0f;
  if (r < F_DIM) {
    size_t row = (size_t)(g * F_DIM + r);
    if (k < F_DIM)                      v = w_ih[row * F_DIM + k];
    else if (k >= FP && k < FP + F_DIM) v = w_hh[row * F_DIM + (k - FP)];
  }
  Wp[idx] = (__bf16)v;
}

__global__ void pack_bias_kernel(const float* __restrict__ b_ih,
                                 const float* __restrict__ b_hh,
                                 float* __restrict__ bias) {
  int idx = blockIdx.x * blockDim.x + threadIdx.x; // < 2048
  int g = idx >> 9, r = idx & (FP - 1);
  bias[idx] = (r < F_DIM) ? (b_ih[g * F_DIM + r] + b_hh[g * F_DIM + r]) : 0.0f;
}

__global__ void zero_state_kernel(float* c1, float* c2, float* h2f0,
                                  __bf16* h1b0, __bf16* h2b0) {
  size_t idx = (size_t)blockIdx.x * blockDim.x + threadIdx.x; // < 256*512
  c1[idx] = 0.0f; c2[idx] = 0.0f; h2f0[idx] = 0.0f;
  h1b0[idx] = (__bf16)0.0f; h2b0[idx] = (__bf16)0.0f;
}

// ---- fused LSTM-cell step --------------------------------------------------
// grid: 512 WGs of 128 threads (16 m-tiles x 32 col-tiles); wave g computes
// gate g's 16x16 f32 tile via 32x v_wmma_f32_16x16x32_bf16, tiles meet in LDS,
// then a fused elementwise phase updates c and h.
// Optional extra WG (blockIdx.x==512): linear head y[:, t_y] from h2 state.
__global__ void lstm_step_kernel(const __bf16* __restrict__ Asrc0, // [B,512] K half 0
                                 const __bf16* __restrict__ Asrc1, // [B,512] K half 1 (prev h)
                                 const __bf16* __restrict__ Wp,    // [4*512*1024]
                                 const float*  __restrict__ bias,  // [2048]
                                 float*        __restrict__ cState,// [B*512], in-place
                                 __bf16*       __restrict__ hOutB, // [B*512]
                                 float*        __restrict__ hOutF, // [B*512] or null
                                 const float*  __restrict__ h2y,   // for y path
                                 const float*  __restrict__ w_lin,
                                 const float*  __restrict__ b_lin,
                                 float*        __restrict__ yOut,  // [B,T] or null
                                 int t_y) {
  if (blockIdx.x >= 512) {            // linear-head workgroup (uniform branch)
    if (t_y >= 0) {
      for (int b = threadIdx.x; b < BATCH; b += blockDim.x) {
        float acc = b_lin[0];
        for (int k = 0; k < F_DIM; ++k)
          acc += h2y[(size_t)b * FP + k] * w_lin[k];
        yOut[(size_t)b * TSTEP + t_y] = acc;
      }
    }
    return;
  }

  __shared__ float gtile[4][16][16];

  const int lane = threadIdx.x & 31;
  const int gq   = threadIdx.x >> 5;   // wave -> gate (0=i,1=f,2=g,3=o)
  const int hs   = lane >> 4;          // lane half select
  const int l15  = lane & 15;
  const int mt   = blockIdx.x & 15;    // batch tile
  const int jt   = blockIdx.x >> 4;    // column tile within a gate
  const int mrow = mt * 16 + l15;
  const int jcol = jt * 16 + l15;

  const __bf16* aRow0 = Asrc0 + (size_t)mrow * FP;
  const __bf16* aRow1 = Asrc1 + (size_t)mrow * FP;
  const __bf16* bRow  = Wp + ((size_t)gq * FP + jcol) * KDIM;

  v8f acc = {};
  // K half 0: x (layer1) or h1_new (layer2)
  #pragma unroll 4
  for (int k0 = 0; k0 < FP; k0 += 32) {
    v16bf a = load_frag16(aRow0 + k0 + hs * 8, aRow0 + k0 + 16 + hs * 8);
    v16bf b = load_frag16(bRow  + k0 + hs * 8, bRow + k0 + 16 + hs * 8);
    acc = __builtin_amdgcn_wmma_f32_16x16x32_bf16(false, a, false, b,
                                                  (short)0, acc, false, false);
  }
  // K half 1: this layer's previous h
  #pragma unroll 4
  for (int k0 = FP; k0 < KDIM; k0 += 32) {
    v16bf a = load_frag16(aRow1 + (k0 - FP) + hs * 8,
                          aRow1 + (k0 - FP) + 16 + hs * 8);
    v16bf b = load_frag16(bRow + k0 + hs * 8, bRow + k0 + 16 + hs * 8);
    acc = __builtin_amdgcn_wmma_f32_16x16x32_bf16(false, a, false, b,
                                                  (short)0, acc, false, false);
  }

  // D layout: VGPR r -> M = r + 8*hs (lanes 0-15 M=0..7, lanes 16-31 M=8..15)
  #pragma unroll
  for (int r = 0; r < 8; ++r)
    gtile[gq][hs * 8 + r][l15] = acc[r];
  __syncthreads();

  // fused cell update: 128 threads cover 256 elements of the 16x16 patch
  for (int e = threadIdx.x; e < 256; e += 128) {
    int ml  = e >> 4, jl = e & 15;
    int col = jt * 16 + jl;            // gate column (0..511, padded tail)
    int row = mt * 16 + ml;            // batch row
    float ig = sigmoidf(gtile[0][ml][jl] + bias[           col]);
    float fg = sigmoidf(gtile[1][ml][jl] + bias[    FP  +  col]);
    float gg = tanhf   (gtile[2][ml][jl] + bias[2 * FP  +  col]);
    float og = sigmoidf(gtile[3][ml][jl] + bias[3 * FP  +  col]);
    size_t idx = (size_t)row * FP + col;
    float cn = fg * cState[idx] + ig * gg;
    float hn = og * tanhf(cn);
    cState[idx] = cn;
    hOutB[idx]  = (__bf16)hn;
    if (hOutF) hOutF[idx] = hn;
  }
}

__global__ void final_y_kernel(const float* __restrict__ h2,
                               const float* __restrict__ w_lin,
                               const float* __restrict__ b_lin,
                               float* __restrict__ y, int t) {
  for (int b = threadIdx.x; b < BATCH; b += blockDim.x) {
    float acc = b_lin[0];
    for (int k = 0; k < F_DIM; ++k)
      acc += h2[(size_t)b * FP + k] * w_lin[k];
    y[(size_t)b * TSTEP + t] = acc;
  }
}

// ---------------------------------------------------------------------------

extern "C" void kernel_launch(void* const* d_in, const int* in_sizes, int n_in,
                              void* d_out, int out_size, void* d_ws, size_t ws_size,
                              hipStream_t stream) {
  const float* x     = (const float*)d_in[0];
  const float* w_ih1 = (const float*)d_in[1];
  const float* w_hh1 = (const float*)d_in[2];
  const float* b_ih1 = (const float*)d_in[3];
  const float* b_hh1 = (const float*)d_in[4];
  const float* w_ih2 = (const float*)d_in[5];
  const float* w_hh2 = (const float*)d_in[6];
  const float* b_ih2 = (const float*)d_in[7];
  const float* b_hh2 = (const float*)d_in[8];
  const float* w_lin = (const float*)d_in[9];
  const float* b_lin = (const float*)d_in[10];
  float* yOut = (float*)d_out;

  // ---- carve workspace (~139 MB) ----
  char* w = (char*)d_ws;
  __bf16* xb = (__bf16*)w;  w += (size_t)TSTEP * BATCH * FP * sizeof(__bf16); // 128 MB
  __bf16* Wp1 = (__bf16*)w; w += (size_t)4 * FP * KDIM * sizeof(__bf16);      // 4 MB
  __bf16* Wp2 = (__bf16*)w; w += (size_t)4 * FP * KDIM * sizeof(__bf16);      // 4 MB
  float* bias1 = (float*)w; w += (size_t)4 * FP * sizeof(float);
  float* bias2 = (float*)w; w += (size_t)4 * FP * sizeof(float);
  const size_t SN = (size_t)BATCH * FP;
  float* c1 = (float*)w;  w += SN * sizeof(float);
  float* c2 = (float*)w;  w += SN * sizeof(float);
  float* h2f[2];
  h2f[0] = (float*)w;  w += SN * sizeof(float);
  h2f[1] = (float*)w;  w += SN * sizeof(float);
  __bf16* h1b[2];
  h1b[0] = (__bf16*)w; w += SN * sizeof(__bf16);
  h1b[1] = (__bf16*)w; w += SN * sizeof(__bf16);
  __bf16* h2b[2];
  h2b[0] = (__bf16*)w; w += SN * sizeof(__bf16);
  h2b[1] = (__bf16*)w; w += SN * sizeof(__bf16);

  // ---- one-time packing ----
  conv_x_kernel<<<dim3((unsigned)((size_t)TSTEP * BATCH * FP / 256)), 256, 0, stream>>>(x, xb);
  pack_w_kernel<<<dim3((unsigned)((size_t)4 * FP * KDIM / 256)), 256, 0, stream>>>(w_ih1, w_hh1, Wp1);
  pack_w_kernel<<<dim3((unsigned)((size_t)4 * FP * KDIM / 256)), 256, 0, stream>>>(w_ih2, w_hh2, Wp2);
  pack_bias_kernel<<<dim3(8), 256, 0, stream>>>(b_ih1, b_hh1, bias1);
  pack_bias_kernel<<<dim3(8), 256, 0, stream>>>(b_ih2, b_hh2, bias2);
  zero_state_kernel<<<dim3((unsigned)(SN / 256)), 256, 0, stream>>>(c1, c2, h2f[0], h1b[0], h2b[0]);

  // ---- recurrence: 2 fused kernels per timestep ----
  for (int t = 0; t < TSTEP; ++t) {
    const int rp = t & 1;       // read parity
    const int wp = rp ^ 1;      // write parity
    // layer 1 (+ linear head of step t-1 in one extra workgroup)
    lstm_step_kernel<<<dim3(512 + (t > 0 ? 1 : 0)), 128, 0, stream>>>(
        xb + (size_t)t * BATCH * FP,   // A half 0 = x_t
        h1b[rp],                       // A half 1 = h1_{t-1}
        Wp1, bias1, c1,
        h1b[wp], nullptr,
        h2f[rp], w_lin, b_lin, yOut, t - 1);
    // layer 2
    lstm_step_kernel<<<dim3(512), 128, 0, stream>>>(
        h1b[wp],                       // A half 0 = h1_t
        h2b[rp],                       // A half 1 = h2_{t-1}
        Wp2, bias2, c2,
        h2b[wp], h2f[wp],
        nullptr, w_lin, b_lin, nullptr, -1);
  }
  // linear head for the last timestep (step 511 wrote parity 0)
  final_y_kernel<<<dim3(1), 128, 0, stream>>>(h2f[0], w_lin, b_lin, yOut, TSTEP - 1);
}
